// LatentODE_82892868813105
// MI455X (gfx1250) — matrix operand order repaired and checked by reading, not compiled, not run on hardware
//
#include <hip/hip_runtime.h>
#include <hip/hip_bf16.h>

// ---------------------------------------------------------------------------
// Latent ODE, fully fused for MI455X (gfx1250, wave32, WMMA 16x16x32 f16).
// Grid: 16 workgroups (one per 16-batch tile), 128 threads (4 waves) each.
// Activations live in LDS as f16 so A-fragments are two ds_load_b128 each
// (per-lane A layout = two contiguous 8-half runs). Weights live in VGPR
// B-fragments for the entire 1024-step scans.
// ---------------------------------------------------------------------------

typedef __attribute__((ext_vector_type(16))) _Float16 v16h;
typedef __attribute__((ext_vector_type(8)))  _Float16 v8h;
typedef __attribute__((ext_vector_type(4)))  _Float16 v4h;
typedef __attribute__((ext_vector_type(8)))  float    v8f;

#define B_  256
#define T_  1024
#define D_  32
#define H_  64
#define L_  32
#define DT0_ 0.4f

#define XSH 72   // f16 buffers: 72 halves = 144 B row stride (36 banks mod 64)
#define GS  196  // f32 gate buffers: 196 floats (4 banks mod 64)

#define WMMA_F16(A, B, C) \
  __builtin_amdgcn_wmma_f32_16x16x32_f16(false, (A), false, (B), (short)0, (C), false, false)

// A fragment (16x32 f16) from row-major f16 LDS: two 16-byte ds_load_b128.
// Per-lane elements: e=0..7 -> K = kbase + 8*(lane>>4) + e
//                    e=8..15 -> K = kbase + 8*(lane>>4) + 16 + (e-8)
__device__ __forceinline__ v16h frag_a_h(const _Float16* sm, int kbase) {
  const int lane = threadIdx.x & 31;
  const int m    = lane & 15;
  const int kh   = kbase + ((lane >> 4) << 3);
  v8h lo = *(const v8h*)(sm + m * XSH + kh);
  v8h hi = *(const v8h*)(sm + m * XSH + kh + 16);
  return __builtin_shufflevector(lo, hi, 0, 1, 2, 3, 4, 5, 6, 7,
                                 8, 9, 10, 11, 12, 13, 14, 15);
}

// B fragment (32x16 f16): B[k][n] = W[n][k]  (computes x @ W^T)
__device__ __forceinline__ v16h frag_b_w(const float* W, int ldk, int nbase,
                                         int kbase, int klim) {
  const int lane = threadIdx.x & 31;
  const int n    = nbase + (lane & 15);
  const int k0   = kbase + ((lane >> 4) << 3);
  v16h b;
#pragma unroll
  for (int e = 0; e < 16; ++e) {
    int k = k0 + e + ((e >> 3) << 3);
    b[e]  = (_Float16)((k < klim) ? W[n * ldk + k] : 0.0f);
  }
  return b;
}

// gru_wi B fragment with permuted K: data cols 0..31 = ys (wi col k+1),
// col 32 = ts (wi col 0), cols 33..63 = zero pad.
__device__ __forceinline__ v16h frag_b_gwi(const float* W, int nbase, int kbase) {
  const int lane = threadIdx.x & 31;
  const int n    = nbase + (lane & 15);
  const int k0   = kbase + ((lane >> 4) << 3);
  v16h b;
#pragma unroll
  for (int e = 0; e < 16; ++e) {
    int k = k0 + e + ((e >> 3) << 3);
    float w = 0.0f;
    if (k < 32)       w = W[n * 33 + k + 1];
    else if (k == 32) w = W[n * 33];
    b[e] = (_Float16)w;
  }
  return b;
}

__device__ __forceinline__ v8f frag_c_splat(float v) {
  v8f c;
#pragma unroll
  for (int r = 0; r < 8; ++r) c[r] = v;
  return c;
}

// C/D layout: VGPR r -> row (r + 8*(lane>>4)), col (lane&15)
__device__ __forceinline__ void store_acc_f32(float* sm, int stride, int nbase, v8f acc) {
  const int lane = threadIdx.x & 31;
  const int n    = nbase + (lane & 15);
  const int mo   = (lane >> 4) << 3;
#pragma unroll
  for (int r = 0; r < 8; ++r) sm[(mo + r) * stride + n] = acc[r];
}
__device__ __forceinline__ void store_acc_f16(_Float16* sm, int nbase, v8f acc) {
  const int lane = threadIdx.x & 31;
  const int n    = nbase + (lane & 15);
  const int mo   = (lane >> 4) << 3;
#pragma unroll
  for (int r = 0; r < 8; ++r) sm[(mo + r) * XSH + n] = (_Float16)acc[r];
}

__device__ __forceinline__ float sigmoidf_(float x) { return 1.0f / (1.0f + __expf(-x)); }
__device__ __forceinline__ float softplusf_(float x) {
  return (x > 20.0f) ? x : __logf(1.0f + __expf(x));
}
__device__ __forceinline__ float fast_tanh(float x) {
  x = fminf(fmaxf(x, -15.0f), 15.0f);
  float e = __expf(2.0f * x);
  return (e - 1.0f) / (e + 1.0f);
}

__global__ __launch_bounds__(128, 1)
void latentode_fused(const float* __restrict__ ts, const float* __restrict__ ys,
                     const float* __restrict__ eps, const float* __restrict__ scale,
                     const float* __restrict__ fw0, const float* __restrict__ fb0,
                     const float* __restrict__ fw1, const float* __restrict__ fb1,
                     const float* __restrict__ fw2, const float* __restrict__ fb2,
                     const float* __restrict__ gwi, const float* __restrict__ gwh,
                     const float* __restrict__ gb,  const float* __restrict__ gbn,
                     const float* __restrict__ hlw, const float* __restrict__ hlb,
                     const float* __restrict__ lw0, const float* __restrict__ lb0,
                     const float* __restrict__ lw1, const float* __restrict__ lb1,
                     const float* __restrict__ lw2, const float* __restrict__ lb2,
                     const float* __restrict__ hdw, const float* __restrict__ hdb,
                     float* __restrict__ out) {
  __shared__ alignas(16) _Float16 sm_x[16 * XSH];  // [ys(0..31) | ts(32) | 0-pad]
  __shared__ alignas(16) _Float16 sm_h[16 * XSH];  // GRU hidden (f16 mirror)
  __shared__ alignas(16) _Float16 sm_y[16 * XSH];  // ODE state (f16 mirror)
  __shared__ alignas(16) _Float16 sm_a[16 * XSH];  // MLP activation
  __shared__ alignas(16) _Float16 sm_t[16 * XSH];  // MLP activation (second)
  __shared__ float sm_y32[16 * 68];                // ODE state master (f32)
  __shared__ float sm_gi[16 * GS];                 // x-path gates / ctx scratch
  __shared__ float sm_gh[16 * GS];                 // h-path gates
  __shared__ float sm_red[32];                     // [0:16)=recon, [16:32)=kl

  const int tid  = threadIdx.x;
  const int wave = tid >> 5;
  const int lane = tid & 31;
  const int b0   = blockIdx.x * 16;

  for (int e = tid; e < 16 * XSH; e += 128) {
    sm_h[e] = (_Float16)0.0f;
    sm_x[e] = (_Float16)0.0f;
  }
  if (tid < 32) sm_red[tid] = 0.0f;

  // ------------- GRU B-fragments + biases, resident across the scan --------
  v16h Bwi[3][2], Bwh[3][2];
  float gbv[3];
#pragma unroll
  for (int j = 0; j < 3; ++j) {
    const int nb = (wave * 3 + j) * 16;   // N tiles 0..11 over 192 gate cols
    gbv[j] = gb[nb + (lane & 15)];
#pragma unroll
    for (int kk = 0; kk < 2; ++kk) {
      Bwi[j][kk] = frag_b_gwi(gwi, nb, kk * 32);
      Bwh[j][kk] = frag_b_w(gwh, 64, nb, kk * 32, 64);
    }
  }
  // gate-update thread mapping: fixed column, rows m0+2i
  const int gc  = tid & 63;
  const int gm0 = tid >> 6;
  const float bnc = gbn[gc];

  // x staging mapping: thread -> (row m, quad q), float4 -> v4h ds_store_b64
  const int xm = tid >> 3;
  const int xq = tid & 7;
  const float* ysrow = ys + (size_t)(b0 + xm) * T_ * D_ + 4 * xq;

  // stage x for t = T-1
  {
    const int t = T_ - 1;
    float4 v = *(const float4*)(ysrow + (size_t)t * D_);
    v4h p; p[0] = (_Float16)v.x; p[1] = (_Float16)v.y;
    p[2] = (_Float16)v.z; p[3] = (_Float16)v.w;
    *(v4h*)(sm_x + xm * XSH + 4 * xq) = p;
    if (tid < 16) sm_x[tid * XSH + 32] = (_Float16)ts[(size_t)(b0 + tid) * T_ + t];
  }
  __syncthreads();

  // ---------------- GRU scan (reversed time), 2 barriers/step --------------
  for (int s = 0; s < T_; ++s) {
    const int t = T_ - 1 - s;
    v16h Ax0 = frag_a_h(sm_x, 0);
    v16h Ax1 = frag_a_h(sm_x, 32);
    v16h Ah0 = frag_a_h(sm_h, 0);
    v16h Ah1 = frag_a_h(sm_h, 32);
#pragma unroll
    for (int j = 0; j < 3; ++j) {
      const int nb = (wave * 3 + j) * 16;
      v8f ai = frag_c_splat(gbv[j]);    // ig = x@Wi^T + b
      ai = WMMA_F16(Ax0, Bwi[j][0], ai);
      ai = WMMA_F16(Ax1, Bwi[j][1], ai);
      v8f ah = frag_c_splat(0.0f);      // hg = h@Wh^T (kept split for n-gate)
      ah = WMMA_F16(Ah0, Bwh[j][0], ah);
      ah = WMMA_F16(Ah1, Bwh[j][1], ah);
      store_acc_f32(sm_gi, GS, nb, ai);
      store_acc_f32(sm_gh, GS, nb, ah);
    }
    __syncthreads();

    // gate nonlinearity + h update; also stage x for next step (sm_x was
    // fully consumed into A fragments before the barrier above).
#pragma unroll
    for (int i = 0; i < 8; ++i) {
      const int m = gm0 + 2 * i;
      float ir = sm_gi[m * GS + gc],        hr = sm_gh[m * GS + gc];
      float iz = sm_gi[m * GS + 64 + gc],   hz = sm_gh[m * GS + 64 + gc];
      float in_ = sm_gi[m * GS + 128 + gc], hn = sm_gh[m * GS + 128 + gc];
      float r = sigmoidf_(ir + hr);
      float z = sigmoidf_(iz + hz);
      float n = fast_tanh(in_ + r * (hn + bnc));
      float h = (float)sm_h[m * XSH + gc];
      sm_h[m * XSH + gc] = (_Float16)(n + z * (h - n));
    }
    if (t > 0) {
      float4 v = *(const float4*)(ysrow + (size_t)(t - 1) * D_);
      v4h p; p[0] = (_Float16)v.x; p[1] = (_Float16)v.y;
      p[2] = (_Float16)v.z; p[3] = (_Float16)v.w;
      *(v4h*)(sm_x + xm * XSH + 4 * xq) = p;
      if (tid < 16) sm_x[tid * XSH + 32] = (_Float16)ts[(size_t)(b0 + tid) * T_ + t - 1];
      if (t > 1) __builtin_prefetch(ysrow + (size_t)(t - 2) * D_, 0, 1);
    }
    __syncthreads();
  }

  // ---------------- context -> latent + KL ---------------------------------
  {
    const int nb = wave * 16;
    v16h Bh0 = frag_b_w(hlw, 64, nb, 0, 64);
    v16h Bh1 = frag_b_w(hlw, 64, nb, 32, 64);
    v16h A0  = frag_a_h(sm_h, 0);
    v16h A1  = frag_a_h(sm_h, 32);
    v8f  c   = frag_c_splat(hlb[nb + (lane & 15)]);
    c = WMMA_F16(A0, Bh0, c);
    c = WMMA_F16(A1, Bh1, c);
    store_acc_f32(sm_gi, GS, nb, c);   // ctx[16][64]
  }
  __syncthreads();

  const float scalev = scale[0];
#pragma unroll
  for (int i = 0; i < 4; ++i) {
    int e = tid + 128 * i;             // 16 rows x 32 latent cols
    int m = e >> 5, c = e & 31;
    float mean   = sm_gi[m * GS + c];
    float logstd = sm_gi[m * GS + 32 + c];
    float std    = __expf(logstd);
    sm_x[m * XSH + c] = (_Float16)(mean + eps[(b0 + m) * L_ + c] * std);
    float klp = 0.5f * (mean * mean + std * std - 2.0f * logstd - 1.0f);
    atomicAdd(&sm_red[16 + m], klp);
  }
  __syncthreads();

  // ---------------- decoder init MLP: latent(32)->64->64->y0(64) -----------
  {
    const int nb = wave * 16;
    v16h B0 = frag_b_w(lw0, 32, nb, 0, 32);
    v16h A0 = frag_a_h(sm_x, 0);       // K=32 exactly (col 32 = stale ts unused)
    v8f  c  = frag_c_splat(lb0[nb + (lane & 15)]);
    c = WMMA_F16(A0, B0, c);
#pragma unroll
    for (int r = 0; r < 8; ++r) c[r] = fmaxf(c[r], 0.0f);
    store_acc_f16(sm_a, nb, c);
  }
  __syncthreads();
  {
    const int nb = wave * 16;
    v16h B0 = frag_b_w(lw1, 64, nb, 0, 64);
    v16h B1 = frag_b_w(lw1, 64, nb, 32, 64);
    v16h A0 = frag_a_h(sm_a, 0);
    v16h A1 = frag_a_h(sm_a, 32);
    v8f  c  = frag_c_splat(lb1[nb + (lane & 15)]);
    c = WMMA_F16(A0, B0, c);
    c = WMMA_F16(A1, B1, c);
#pragma unroll
    for (int r = 0; r < 8; ++r) c[r] = fmaxf(c[r], 0.0f);
    store_acc_f16(sm_t, nb, c);
  }
  __syncthreads();
  {
    const int nb = wave * 16;
    v16h B0 = frag_b_w(lw2, 64, nb, 0, 64);
    v16h B1 = frag_b_w(lw2, 64, nb, 32, 64);
    v16h A0 = frag_a_h(sm_t, 0);
    v16h A1 = frag_a_h(sm_t, 32);
    v8f  c  = frag_c_splat(lb2[nb + (lane & 15)]);
    c = WMMA_F16(A0, B0, c);
    c = WMMA_F16(A1, B1, c);
    // y0: master f32 + f16 mirror
    const int n  = nb + (lane & 15);
    const int mo = (lane >> 4) << 3;
#pragma unroll
    for (int r = 0; r < 8; ++r) {
      sm_y32[(mo + r) * 68 + n] = c[r];
      sm_y[(mo + r) * XSH + n]  = (_Float16)c[r];
    }
  }
  __syncthreads();

  // ---------------- Euler scan + recon -------------------------------------
  v16h Bf0[2], Bf1[2], Bf2[2], Bhd[2];
  float bf0v, bf1v, bf2v, hdbv;
  {
    const int nb  = wave * 16;
    const int nb2 = (wave & 1) * 16;   // hd proj (D=32): tiles for waves 0,1
    bf0v = fb0[nb + (lane & 15)];
    bf1v = fb1[nb + (lane & 15)];
    bf2v = fb2[nb + (lane & 15)];
    hdbv = hdb[nb2 + (lane & 15)];
#pragma unroll
    for (int kk = 0; kk < 2; ++kk) {
      Bf0[kk] = frag_b_w(fw0, 64, nb,  kk * 32, 64);
      Bf1[kk] = frag_b_w(fw1, 64, nb,  kk * 32, 64);
      Bf2[kk] = frag_b_w(fw2, 64, nb,  kk * 32, 64);
      Bhd[kk] = frag_b_w(hdw, 64, nb2, kk * 32, 64);
    }
  }
  float racc[8];
#pragma unroll
  for (int r = 0; r < 8; ++r) racc[r] = 0.0f;
  const size_t ysb = (size_t)b0 * T_ * D_;

  for (int t = 0; t < T_; ++t) {
    const int nb = wave * 16;
    // layer 1: a = softplus(y @ fw0^T + fb0)
    {
      v16h A0 = frag_a_h(sm_y, 0);
      v16h A1 = frag_a_h(sm_y, 32);
      v8f  c  = frag_c_splat(bf0v);
      c = WMMA_F16(A0, Bf0[0], c);
      c = WMMA_F16(A1, Bf0[1], c);
#pragma unroll
      for (int r = 0; r < 8; ++r) c[r] = softplusf_(c[r]);
      store_acc_f16(sm_a, nb, c);
    }
    __syncthreads();
    // layer 2
    {
      v16h A0 = frag_a_h(sm_a, 0);
      v16h A1 = frag_a_h(sm_a, 32);
      v8f  c  = frag_c_splat(bf1v);
      c = WMMA_F16(A0, Bf1[0], c);
      c = WMMA_F16(A1, Bf1[1], c);
#pragma unroll
      for (int r = 0; r < 8; ++r) c[r] = softplusf_(c[r]);
      store_acc_f16(sm_t, nb, c);
    }
    __syncthreads();
    // layer 3 + Euler update: y += DT0 * scale * tanh(.)
    {
      v16h A0 = frag_a_h(sm_t, 0);
      v16h A1 = frag_a_h(sm_t, 32);
      v8f  c  = frag_c_splat(bf2v);
      c = WMMA_F16(A0, Bf2[0], c);
      c = WMMA_F16(A1, Bf2[1], c);
      const int n  = nb + (lane & 15);
      const int mo = (lane >> 4) << 3;
#pragma unroll
      for (int r = 0; r < 8; ++r) {
        float yn = sm_y32[(mo + r) * 68 + n] + DT0_ * scalev * fast_tanh(c[r]);
        sm_y32[(mo + r) * 68 + n] = yn;
        sm_y[(mo + r) * XSH + n]  = (_Float16)yn;
      }
    }
    __syncthreads();
    // pred = y_new @ hd_w^T + hd_b ; recon += (ys - pred)^2  (waves 0,1)
    if (wave < 2) {
      const int nb2 = wave * 16;
      v16h A0 = frag_a_h(sm_y, 0);
      v16h A1 = frag_a_h(sm_y, 32);
      v8f  c  = frag_c_splat(hdbv);
      c = WMMA_F16(A0, Bhd[0], c);
      c = WMMA_F16(A1, Bhd[1], c);
      const int n  = nb2 + (lane & 15);
      const int mo = (lane >> 4) << 3;
#pragma unroll
      for (int r = 0; r < 8; ++r) {
        const int m = mo + r;
        float d = ys[ysb + (size_t)m * T_ * D_ + (size_t)t * D_ + n] - c[r];
        racc[r] += d * d;
      }
      if (t + 1 < T_)
        __builtin_prefetch(&ys[ysb + (size_t)mo * T_ * D_ + (size_t)(t + 1) * D_ + n], 0, 1);
    }
  }

  // ---------------- reductions + output ------------------------------------
  if (wave < 2) {
    const int mo = (lane >> 4) << 3;
#pragma unroll
    for (int r = 0; r < 8; ++r) atomicAdd(&sm_red[mo + r], racc[r]);
  }
  __syncthreads();
  if (tid < 16) out[b0 + tid] = 0.5f * sm_red[tid] + sm_red[16 + tid];
}

extern "C" void kernel_launch(void* const* d_in, const int* in_sizes, int n_in,
                              void* d_out, int out_size, void* d_ws, size_t ws_size,
                              hipStream_t stream) {
  (void)in_sizes; (void)n_in; (void)out_size; (void)d_ws; (void)ws_size;
  latentode_fused<<<dim3(B_ / 16), dim3(128), 0, stream>>>(
      (const float*)d_in[0],  (const float*)d_in[1],  (const float*)d_in[2],
      (const float*)d_in[3],  (const float*)d_in[4],  (const float*)d_in[5],
      (const float*)d_in[6],  (const float*)d_in[7],  (const float*)d_in[8],
      (const float*)d_in[9],  (const float*)d_in[10], (const float*)d_in[11],
      (const float*)d_in[12], (const float*)d_in[13], (const float*)d_in[14],
      (const float*)d_in[15], (const float*)d_in[16], (const float*)d_in[17],
      (const float*)d_in[18], (const float*)d_in[19], (const float*)d_in[20],
      (const float*)d_in[21], (const float*)d_in[22], (const float*)d_in[23],
      (float*)d_out);
}
